// SVAE_86646670230084
// MI455X (gfx1250) — compile-verified, matrix operand order checked by reading, not run on hardware
//
#include <hip/hip_runtime.h>
#include <hip/hip_bf16.h>

// ---------------------------------------------------------------------------
// SVAE forward for MI455X (gfx1250, wave32, WMMA).
// - Heavy GEMMs: v_wmma_f32_16x16x32_bf16
// - Decoder stage-2 (82 GFLOP of the ~86 total): LDS-staged, double-buffered
//   block GEMM fed by global_load_async_to_lds_b128 (ASYNCcnt).
// - Per-(n,k) 64x64 linear algebra: LDS Cholesky blocks.
// ---------------------------------------------------------------------------

#define NB   512      // batch
#define DD   64       // latent dim
#define KK   10       // mixture comps
#define SS   10       // samples
#define INDIM 784
#define HID  512
#define INP  800      // 784 padded to multiple of 32 for WMMA K-loop
#define LOG2PI_F 1.8378770664093453f

typedef __attribute__((ext_vector_type(16))) __bf16          v16bf;
typedef __attribute__((ext_vector_type(8)))  float           v8f;
typedef __attribute__((ext_vector_type(8)))  unsigned short  v8u;
typedef __attribute__((ext_vector_type(16))) unsigned short  v16u;

#if __has_builtin(__builtin_amdgcn_global_load_async_to_lds_b128) && \
    __has_builtin(__builtin_amdgcn_s_wait_asynccnt)
#define ASYNC_LDS 1
// Builtin wants: (v4i addrspace(1)*, v4i addrspace(3)*, imm offset, imm cpol)
// where v4i is a GCC-style vector_size(16) int (per the clang diagnostic).
typedef int v4i_t __attribute__((vector_size(4 * sizeof(int))));
typedef __attribute__((address_space(1))) v4i_t* gv4_t;
typedef __attribute__((address_space(3))) v4i_t* lv4_t;
#endif

// float -> bf16 (round-to-nearest-even) without relying on __bf16 arithmetic
__device__ __forceinline__ unsigned short f2bf(float f) {
    unsigned u = __builtin_bit_cast(unsigned, f);
    u += 0x7FFFu + ((u >> 16) & 1u);
    return (unsigned short)(u >> 16);
}

__device__ __forceinline__ v16bf mk_frag(v8u lo, v8u hi) {
    v16u v = __builtin_shufflevector(lo, hi, 0,1,2,3,4,5,6,7,8,9,10,11,12,13,14,15);
    return __builtin_bit_cast(v16bf, v);
}

// ---- tiny counter-based RNG (hash -> Box-Muller) --------------------------
__device__ __forceinline__ unsigned pcg(unsigned v) {
    v = v * 747796405u + 2891336453u;
    v = ((v >> ((v >> 28) + 4u)) ^ v) * 277803737u;
    return (v >> 22) ^ v;
}
__device__ __forceinline__ float u01(unsigned s) {
    return (float)(pcg(s) >> 8) * (1.0f / 16777216.0f) + 5.9604645e-8f;
}
__device__ __forceinline__ float gaussian(unsigned s) {
    float a = u01(s * 2u + 0x9E3779B9u);
    float b = u01(s * 2u + 0x85EBCA6Bu);
    return sqrtf(-2.0f * logf(a)) * cosf(6.28318530718f * b);
}

__device__ __forceinline__ float act_apply(float v, int act) {
    if (act == 1)      v = tanhf(v);
    else if (act == 2) v = (v > 20.0f ? v : log1pf(expf(v))) + 1e-6f;
    return v;
}

// ===========================================================================
// bf16 conversion kernels (row-major pad, and transpose-to-[N,Kp] for B)
// ===========================================================================
__global__ void cvt_pad(const float* __restrict__ src, unsigned short* __restrict__ dst,
                        int R, int C, int Cp) {
    long i = (long)blockIdx.x * 256 + threadIdx.x;
    long tot = (long)R * Cp;
    if (i >= tot) return;
    int r = (int)(i / Cp), c = (int)(i % Cp);
    dst[i] = f2bf(c < C ? src[(long)r * C + c] : 0.0f);
}

__global__ void cvt_T_pad(const float* __restrict__ src /*[K,N]*/,
                          unsigned short* __restrict__ dst /*[N,Kp]*/,
                          int Kd, int N, int Kp) {
    long i = (long)blockIdx.x * 256 + threadIdx.x;
    long tot = (long)N * Kp;
    if (i >= tot) return;
    int n = (int)(i / Kp), k = (int)(i % Kp);
    dst[i] = f2bf(k < Kd ? src[(long)k * N + n] : 0.0f);
}

// ===========================================================================
// Generic direct-load WMMA GEMM (one 16x16 tile per wave); used for the small
// GEMMs where operands live in L2 anyway.
// act: 0=none, 1=tanh, 2=softplus+1e-6
// ===========================================================================
__global__ __launch_bounds__(256) void wmma_gemm(
    const unsigned short* __restrict__ A,
    const unsigned short* __restrict__ Bt,
    const float* __restrict__ bias,
    float* __restrict__ Cf,
    unsigned short* __restrict__ Cb,
    int M, int N, int Kp, int act)
{
    const int lane = threadIdx.x & 31;
    const int wave = threadIdx.x >> 5;
    const int gw = blockIdx.x * 8 + wave;
    const int tilesN = N >> 4;
    const int tm = gw / tilesN;
    const int tn = gw % tilesN;
    if (tm >= (M >> 4)) return;

    const int r  = lane & 15;
    const int kb = (lane >> 4) * 8;
    const unsigned short* arow = A  + (long)(tm * 16 + r) * Kp;
    const unsigned short* brow = Bt + (long)(tn * 16 + r) * Kp;

    v8f acc = {};
    for (int kk = 0; kk < Kp; kk += 32) {
        v8u a0 = *(const v8u*)(arow + kk + kb);
        v8u a1 = *(const v8u*)(arow + kk + kb + 16);
        v8u b0 = *(const v8u*)(brow + kk + kb);
        v8u b1 = *(const v8u*)(brow + kk + kb + 16);
        __builtin_prefetch(arow + kk + 256, 0, 1);   // global_prefetch_b8
        __builtin_prefetch(brow + kk + 256, 0, 1);
        acc = __builtin_amdgcn_wmma_f32_16x16x32_bf16(
                  false, mk_frag(a0, a1), false, mk_frag(b0, b1),
                  (short)0, acc, false, false);
    }

    const int n  = tn * 16 + r;
    const int m0 = tm * 16 + (lane >> 4) * 8;
    const float bv_ = bias ? bias[n] : 0.0f;
    for (int i = 0; i < 8; i++) {
        float v = act_apply(acc[i] + bv_, act);
        long idx = (long)(m0 + i) * N + n;
        if (Cf) Cf[idx] = v;
        if (Cb) Cb[idx] = f2bf(v);
    }
}

// ===========================================================================
// Decoder block GEMM: 128(M) x 112(N) block tile, BK=32, double-buffered LDS
// staged via global_load_async_to_lds_b128 (fallback: sync copy). 8 waves,
// each wave owns a 16-row strip and 7 accumulators (7 WMMAs / K-step).
// Per stage each wave issues exactly 4 async-load instructions (2 A + 2 B),
// so s_wait_asynccnt 4 after issuing the next stage retires the current one.
// ===========================================================================
#define BMD 128
#define BND 112

__global__ __launch_bounds__(256) void wmma_gemm_dec(
    const unsigned short* __restrict__ A,   // [M,Kp] bf16
    const unsigned short* __restrict__ Bt,  // [N,Kp] bf16
    const float* __restrict__ bias,
    float* __restrict__ Cf,
    int M, int N, int Kp, int act)
{
    __shared__ unsigned short sA[2][BMD * 32];
    __shared__ unsigned short sB[2][BND * 32];

    const int tid  = threadIdx.x;
    const int lane = tid & 31;
    const int wave = tid >> 5;
    const int nbx  = N / BND;                // 7
    const long m0  = (long)(blockIdx.x / nbx) * BMD;
    const int  n0  = (blockIdx.x % nbx) * BND;

    // async chunk geometry: one wave-instruction moves 8 rows x 32 cols (512B)
    const int rA = lane >> 2;                // 0..7  row within chunk
    const int cA = (lane & 3) * 8;           // 0,8,16,24 col

    auto stageA = [&](int buf, int kk) {
        for (int i = 0; i < 2; i++) {
            int q   = wave * 2 + i;          // 0..15 -> rows 8q..8q+7 (128 rows)
            int row = q * 8 + rA;
            const unsigned short* g = A + (m0 + row) * Kp + kk + cA;
            unsigned short* l = &sA[buf][row * 32 + cA];
#ifdef ASYNC_LDS
            __builtin_amdgcn_global_load_async_to_lds_b128(
                (gv4_t)(const void*)g, (lv4_t)(void*)l, 0, 0);
#else
            *(v8u*)l = *(const v8u*)g;
#endif
        }
    };
    auto stageB = [&](int buf, int kk) {
        for (int i = 0; i < 2; i++) {
            int q = wave * 2 + i;            // 112 rows -> 14 chunks; clamp dups
            if (q > 13) q = 13;
            int row = q * 8 + rA;
            const unsigned short* g = Bt + (long)(n0 + row) * Kp + kk + cA;
            unsigned short* l = &sB[buf][row * 32 + cA];
#ifdef ASYNC_LDS
            __builtin_amdgcn_global_load_async_to_lds_b128(
                (gv4_t)(const void*)g, (lv4_t)(void*)l, 0, 0);
#else
            *(v8u*)l = *(const v8u*)g;
#endif
        }
    };

    v8f acc[7];
    for (int t = 0; t < 7; t++) acc[t] = (v8f){};

    const int r  = lane & 15;
    const int kb = (lane >> 4) * 8;

    stageA(0, 0);
    stageB(0, 0);

    int it = 0;
    for (int kk = 0; kk < Kp; kk += 32, it++) {
        const int cur = it & 1;
        const bool more = (kk + 32) < Kp;
        if (more) { stageA(cur ^ 1, kk + 32); stageB(cur ^ 1, kk + 32); }
#ifdef ASYNC_LDS
        if (more) __builtin_amdgcn_s_wait_asynccnt(4);
        else      __builtin_amdgcn_s_wait_asynccnt(0);
#endif
        __syncthreads();                      // current stage visible to all

        const unsigned short* abase = &sA[cur][(wave * 16 + r) * 32];
        v8u a0 = *(const v8u*)(abase + kb);           // ds_load_b128
        v8u a1 = *(const v8u*)(abase + kb + 16);
        v16bf av = mk_frag(a0, a1);
        for (int t = 0; t < 7; t++) {
            const unsigned short* bbase = &sB[cur][(t * 16 + r) * 32];
            v8u b0 = *(const v8u*)(bbase + kb);
            v8u b1 = *(const v8u*)(bbase + kb + 16);
            acc[t] = __builtin_amdgcn_wmma_f32_16x16x32_bf16(
                         false, av, false, mk_frag(b0, b1),
                         (short)0, acc[t], false, false);
        }
        __syncthreads();                      // done reading before restage
    }

    const int mrow0 = (int)m0 + wave * 16 + (lane >> 4) * 8;
    for (int t = 0; t < 7; t++) {
        const int n = n0 + t * 16 + r;
        const float bb = bias ? bias[n] : 0.0f;
        for (int i = 0; i < 8; i++) {
            float v = act_apply(acc[t][i] + bb, act);
            Cf[(long)(mrow0 + i) * N + n] = v;
        }
    }
}

// ===========================================================================
// GMM precompute: per k build L_k (softplus diag), J_k = L L^T, Sigma_k = J^-1
// ===========================================================================
__global__ void gmm_precompute(const float* __restrict__ Lraw,
                               float* __restrict__ Jg, float* __restrict__ Sg) {
    const int k = blockIdx.x, tid = threadIdx.x;  // 64 threads
    __shared__ float Lm[64][65];
    for (int j = 0; j < 64; j++) {
        float v = Lraw[((long)k * 64 + tid) * 64 + j];
        if (j > tid)       v = 0.0f;
        else if (j == tid) v = (v > 20.0f ? v : log1pf(expf(v)));
        Lm[tid][j] = v;
    }
    __syncthreads();
    for (int j = 0; j < 64; j++) {
        int m = tid < j ? tid : j;
        float s = 0.0f;
        for (int p = 0; p <= m; p++) s += Lm[tid][p] * Lm[j][p];
        Jg[((long)k * 64 + tid) * 64 + j] = s;
    }
    float col[64];
    for (int i = 0; i < 64; i++) col[i] = (i == tid) ? 1.0f : 0.0f;
    for (int j = 0; j < 64; j++) {
        col[j] /= Lm[j][j];
        for (int i = j + 1; i < 64; i++) col[i] -= Lm[i][j] * col[j];
    }
    for (int j = 63; j >= 0; --j) {
        col[j] /= Lm[j][j];
        for (int i = 0; i < j; i++) col[i] -= Lm[j][i] * col[j];
    }
    for (int i = 0; i < 64; i++) Sg[((long)k * 64 + i) * 64 + tid] = col[i];
}

__global__ void logpi_kernel(const float* __restrict__ pir, float* __restrict__ lp) {
    if (threadIdx.x == 0) {
        float mx = -1e30f;
        for (int k = 0; k < KK; k++) mx = fmaxf(mx, pir[k]);
        float se = 0.0f;
        for (int k = 0; k < KK; k++) se += expf(pir[k] - mx);
        float l = mx + logf(se);
        for (int k = 0; k < KK; k++) lp[k] = pir[k] - l;
    }
}

__global__ void reciprocal_kernel(const float* __restrict__ v, float* __restrict__ d, int n) {
    int i = blockIdx.x * 256 + threadIdx.x;
    if (i < n) d[i] = 1.0f / v[i];
}

// ===========================================================================
// LDS 64x64 Cholesky + triangular solves (64 threads, row-per-thread)
// ===========================================================================
__device__ void chol64(float (*A)[65], int tid) {
    for (int j = 0; j < 64; j++) {
        __syncthreads();
        if (tid == j) A[j][j] = sqrtf(fmaxf(A[j][j], 1e-20f));
        __syncthreads();
        if (tid > j) A[tid][j] /= A[j][j];
        __syncthreads();
        if (tid > j) {
            float ljt = A[tid][j];
            for (int c = j + 1; c <= tid; c++) A[tid][c] -= ljt * A[c][j];
        }
    }
    __syncthreads();
}
__device__ void fwd64(float (*L)[65], float* b, int tid) {  // L y = b
    for (int j = 0; j < 64; j++) {
        __syncthreads();
        if (tid == j) b[j] /= L[j][j];
        __syncthreads();
        if (tid > j) b[tid] -= L[tid][j] * b[j];
    }
    __syncthreads();
}
__device__ void bwd64(float (*L)[65], float* b, int tid) {  // L^T x = b
    for (int j = 63; j >= 0; --j) {
        __syncthreads();
        if (tid == j) b[j] /= L[j][j];
        __syncthreads();
        if (tid < j) b[tid] -= L[j][tid] * b[j];
    }
    __syncthreads();
}

// ===========================================================================
// Posterior: one block per (n,k).
// A = diag(d) Sigma_k diag(d) + diag(d);  P = diag(d) + J_k
// s' = 1^T P^{-1} mu_k ;  w_eta1 = s' d ;  x_mean = s' A^{-1} d
// ===========================================================================
__global__ __launch_bounds__(64) void posterior_kernel(
    const float* __restrict__ dvec, const float* __restrict__ mu,
    const float* __restrict__ muk,  const float* __restrict__ Jg,
    const float* __restrict__ Sg,   const float* __restrict__ logpi,
    float* __restrict__ logprob, float* __restrict__ xs,
    unsigned short* __restrict__ xsb)
{
    const int b = blockIdx.x, n = b / KK, k = b % KK, tid = threadIdx.x;
    __shared__ float Mat[64][65];
    __shared__ float d_[64], mu_[64], w_[64], xm_[64], red_[64];
    __shared__ float E[SS][64];
    __shared__ float sc[8];

    d_[tid]  = dvec[(long)n * 64 + tid];
    mu_[tid] = mu[(long)n * 64 + tid];
    __syncthreads();

    // ---- P = diag(d) + J_k ; solve P t = mu_k ; s' = sum(t)
    for (int j = 0; j < 64; j++)
        Mat[tid][j] = Jg[((long)k * 64 + tid) * 64 + j] + (j == tid ? d_[tid] : 0.0f);
    __syncthreads();
    chol64(Mat, tid);
    w_[tid] = muk[(long)k * 64 + tid];
    fwd64(Mat, w_, tid);
    bwd64(Mat, w_, tid);
    red_[tid] = w_[tid];
    __syncthreads();
    if (tid == 0) { float s = 0; for (int i = 0; i < 64; i++) s += red_[i]; sc[0] = s; }

    // ---- A = diag(d) Sigma diag(d) + diag(d)
    for (int j = 0; j < 64; j++)
        Mat[tid][j] = d_[tid] * Sg[((long)k * 64 + tid) * 64 + j] * d_[j]
                      + (j == tid ? d_[tid] : 0.0f);
    __syncthreads();
    {
        float r = 0;
        for (int j = 0; j < 64; j++) r += Mat[tid][j] * mu_[j];
        red_[tid] = r * mu_[tid];
    }
    __syncthreads();
    if (tid == 0) { float s = 0; for (int i = 0; i < 64; i++) s += red_[i]; sc[1] = s; }
    __syncthreads();

    chol64(Mat, tid);
    red_[tid] = logf(Mat[tid][tid]);
    __syncthreads();
    if (tid == 0) { float s = 0; for (int i = 0; i < 64; i++) s += red_[i]; sc[2] = s; }

    // ---- z = A^{-1} d
    w_[tid] = d_[tid];
    fwd64(Mat, w_, tid);
    bwd64(Mat, w_, tid);
    red_[tid] = d_[tid] * w_[tid];
    __syncthreads();
    if (tid == 0) { float s = 0; for (int i = 0; i < 64; i++) s += red_[i]; sc[3] = s; }
    red_[tid] = d_[tid] * mu_[tid];
    __syncthreads();
    if (tid == 0) { float s = 0; for (int i = 0; i < 64; i++) s += red_[i]; sc[4] = s; }
    __syncthreads();

    const float sp = sc[0];
    xm_[tid] = sp * w_[tid];
    if (tid == 0) {
        float lp = sp * sc[4] - 0.5f * sc[1] - 0.5f * sp * sp * sc[3]
                   + sc[2] - 32.0f * LOG2PI_F + logpi[k];
        logprob[(long)n * KK + k] = lp;
    }

    // ---- samples: eps ~ N(0,1), solve L^T e = eps, x = x_mean + e
    unsigned base = (unsigned)(((n * KK + k) * SS) * 64);
    for (int si = 0; si < SS; si++)
        E[si][tid] = gaussian(base + (unsigned)(si * 64 + tid));
    __syncthreads();
    if (tid < SS) {
        for (int j = 63; j >= 0; --j) {
            float v = E[tid][j];
            for (int i = j + 1; i < 64; i++) v -= Mat[i][j] * E[tid][i];
            E[tid][j] = v / Mat[j][j];
        }
    }
    __syncthreads();
    for (int si = 0; si < SS; si++) {
        float x = xm_[tid] + E[si][tid];
        long idx = ((long)(n * KK + k) * SS + si) * 64 + tid;
        xs[idx]  = x;
        xsb[idx] = f2bf(x);
    }
}

// ===========================================================================
// Finalize: log-softmax + categorical pick (s=0) + x_subsamples gather
// ===========================================================================
__global__ __launch_bounds__(64) void finalize_kernel(
    const float* __restrict__ logprob, const float* __restrict__ xs,
    float* __restrict__ out_logz, float* __restrict__ out_xsub)
{
    const int n = blockIdx.x, tid = threadIdx.x;
    __shared__ float lp[KK];
    __shared__ int kc;
    if (tid < KK) lp[tid] = logprob[(long)n * KK + tid];
    __syncthreads();
    if (tid == 0) {
        float mx = lp[0];
        for (int k = 1; k < KK; k++) mx = fmaxf(mx, lp[k]);
        float se = 0;
        for (int k = 0; k < KK; k++) se += expf(lp[k] - mx);
        float lse = mx + logf(se);
        for (int k = 0; k < KK; k++) lp[k] -= lse;
        float u = u01((unsigned)(n * 7919 + 13));
        float c = 0; int sel = KK - 1;
        for (int k = 0; k < KK; k++) { c += expf(lp[k]); if (u <= c) { sel = k; break; } }
        kc = sel;
    }
    __syncthreads();
    if (tid < KK) out_logz[(long)n * KK + tid] = lp[tid];
    out_xsub[(long)n * 64 + tid] = xs[((long)(n * KK + kc) * SS + 0) * 64 + tid];
}

// ===========================================================================
// Host launcher
// ===========================================================================
static inline size_t alup(size_t x) { return (x + 255) & ~(size_t)255; }

extern "C" void kernel_launch(void* const* d_in, const int* in_sizes, int n_in,
                              void* d_out, int out_size, void* d_ws, size_t ws_size,
                              hipStream_t stream) {
    const float* y       = (const float*)d_in[0];
    const float* enc_W1  = (const float*)d_in[1];
    const float* enc_b1  = (const float*)d_in[2];
    const float* enc_Wmu = (const float*)d_in[3];
    const float* enc_bmu = (const float*)d_in[4];
    const float* enc_Wv  = (const float*)d_in[5];
    const float* enc_bv  = (const float*)d_in[6];
    const float* mu_k    = (const float*)d_in[7];
    const float* L_k_raw = (const float*)d_in[8];
    const float* pi_raw  = (const float*)d_in[9];
    const float* dec_W1  = (const float*)d_in[10];
    const float* dec_b1  = (const float*)d_in[11];
    const float* dec_Wmu = (const float*)d_in[12];
    const float* dec_bmu = (const float*)d_in[13];
    const float* dec_Wv  = (const float*)d_in[14];
    const float* dec_bv  = (const float*)d_in[15];

    char* w = (char*)d_ws;
    size_t o = 0;
    auto take = [&](size_t bytes) { size_t r = o; o = alup(o + bytes); return r; };

    unsigned short* y_b    = (unsigned short*)(w + take((size_t)NB * INP * 2));
    unsigned short* W1t    = (unsigned short*)(w + take((size_t)HID * INP * 2));
    unsigned short* Wmut   = (unsigned short*)(w + take((size_t)DD * HID * 2));
    unsigned short* Wvt    = (unsigned short*)(w + take((size_t)DD * HID * 2));
    unsigned short* dW1t   = (unsigned short*)(w + take((size_t)HID * DD * 2));
    unsigned short* dWmut  = (unsigned short*)(w + take((size_t)INDIM * HID * 2));
    unsigned short* dWvt   = (unsigned short*)(w + take((size_t)INDIM * HID * 2));
    unsigned short* h_bf   = (unsigned short*)(w + take((size_t)NB * HID * 2));
    float*          mu_w   = (float*)(w + take((size_t)NB * DD * 4));
    float*          var_w  = (float*)(w + take((size_t)NB * DD * 4));
    float*          d_w    = (float*)(w + take((size_t)NB * DD * 4));
    float*          J_w    = (float*)(w + take((size_t)KK * DD * DD * 4));
    float*          S_w    = (float*)(w + take((size_t)KK * DD * DD * 4));
    float*          lpi_w  = (float*)(w + take(64 * 4));
    float*          lprob  = (float*)(w + take((size_t)NB * KK * 4));
    float*          xs_w   = (float*)(w + take((size_t)NB * KK * SS * DD * 4));
    unsigned short* xsb_w  = (unsigned short*)(w + take((size_t)NB * KK * SS * DD * 2));
    unsigned short* hd_bf  = (unsigned short*)(w + take((size_t)NB * KK * SS * HID * 2));
    (void)ws_size; (void)n_in; (void)in_sizes; (void)out_size;

    float* out      = (float*)d_out;
    float* out_mean = out;                                    // [512,10,10,784]
    float* out_var  = out + (size_t)NB * KK * SS * INDIM;     // [512,10,10,784]
    float* out_xsub = out_var + (size_t)NB * KK * SS * INDIM; // [512,64]
    float* out_logz = out_xsub + (size_t)NB * DD;             // [512,10]

    // ---- bf16 conversions
    cvt_pad  <<<(NB * INP + 255) / 256, 256, 0, stream>>>(y, y_b, NB, INDIM, INP);
    cvt_T_pad<<<(HID * INP + 255) / 256, 256, 0, stream>>>(enc_W1, W1t, INDIM, HID, INP);
    cvt_T_pad<<<(DD * HID + 255) / 256, 256, 0, stream>>>(enc_Wmu, Wmut, HID, DD, HID);
    cvt_T_pad<<<(DD * HID + 255) / 256, 256, 0, stream>>>(enc_Wv, Wvt, HID, DD, HID);
    cvt_T_pad<<<(HID * DD + 255) / 256, 256, 0, stream>>>(dec_W1, dW1t, DD, HID, DD);
    cvt_T_pad<<<(INDIM * HID + 255) / 256, 256, 0, stream>>>(dec_Wmu, dWmut, HID, INDIM, HID);
    cvt_T_pad<<<(INDIM * HID + 255) / 256, 256, 0, stream>>>(dec_Wv, dWvt, HID, INDIM, HID);

    // ---- GMM precompute
    gmm_precompute<<<KK, 64, 0, stream>>>(L_k_raw, J_w, S_w);
    logpi_kernel<<<1, 64, 0, stream>>>(pi_raw, lpi_w);

    // ---- encoder
    {
        int tiles = (NB / 16) * (HID / 16);   // 1024
        wmma_gemm<<<tiles / 8, 256, 0, stream>>>(y_b, W1t, enc_b1, nullptr, h_bf,
                                                 NB, HID, INP, 1);
    }
    {
        int tiles = (NB / 16) * (DD / 16);    // 128
        wmma_gemm<<<tiles / 8, 256, 0, stream>>>(h_bf, Wmut, enc_bmu, mu_w, nullptr,
                                                 NB, DD, HID, 0);
        wmma_gemm<<<tiles / 8, 256, 0, stream>>>(h_bf, Wvt, enc_bv, var_w, nullptr,
                                                 NB, DD, HID, 2);
    }
    reciprocal_kernel<<<(NB * DD + 255) / 256, 256, 0, stream>>>(var_w, d_w, NB * DD);

    // ---- posterior per (n,k)
    posterior_kernel<<<NB * KK, 64, 0, stream>>>(d_w, mu_w, mu_k, J_w, S_w, lpi_w,
                                                 lprob, xs_w, xsb_w);

    // ---- log-softmax + categorical subsample
    finalize_kernel<<<NB, 64, 0, stream>>>(lprob, xs_w, out_logz, out_xsub);

    // ---- decoder
    const int M2 = NB * KK * SS;              // 51200
    {
        int tiles = (M2 / 16) * (HID / 16);   // 102400
        wmma_gemm<<<tiles / 8, 256, 0, stream>>>(xsb_w, dW1t, dec_b1, nullptr, hd_bf,
                                                 M2, HID, DD, 1);
    }
    {
        int blocks = (M2 / BMD) * (INDIM / BND);  // 400 * 7 = 2800
        wmma_gemm_dec<<<blocks, 256, 0, stream>>>(hd_bf, dWmut, dec_bmu, out_mean,
                                                  M2, INDIM, HID, 0);
        wmma_gemm_dec<<<blocks, 256, 0, stream>>>(hd_bf, dWvt, dec_bv, out_var,
                                                  M2, INDIM, HID, 2);
    }
}
